// CausalAttentionModule_30270929502635
// MI455X (gfx1250) — matrix operand order repaired and checked by reading, not compiled, not run on hardware
//
#include <hip/hip_runtime.h>
#include <math.h>

typedef _Float16 half_t;
typedef _Float16 v16h __attribute__((ext_vector_type(16)));
typedef float    v8f  __attribute__((ext_vector_type(8)));

#define WMMA(A, B, C) __builtin_amdgcn_wmma_f32_16x16x32_f16( \
    false, (A), false, (B), (short)0, (C), false, false)

// Compiler-only reordering fence. Each wave communicates exclusively with its
// own lanes through its private LDS slice; CDNA5 LDS ops from one wave are
// processed in order (DScnt in-order), so no s_barrier / hardware wait is
// needed — only a compiler fence to keep program order of LDS accesses.
#define WAVE_LDS_FENCE() asm volatile("" ::: "memory")

// ---------------------------------------------------------------------------
// WMMA operand packing helpers (wave32, 16x16x32 f16)
//
// A-operand (16x32): row = lane&15; lane<16 holds K 0..15 in the 16 halves,
// lane>=16 holds K 16..31 (contiguous k = 16h + s, this file's order).
// B-operand (32x16): lane gives col n=lane&15; lane<16 holds K 0..15 in the
// 16 halves, lane>=16 holds K 16..31 (same contiguous order as load_a).
// ---------------------------------------------------------------------------

__device__ inline v16h load_a(const half_t* p, int row0, int stride, int kOff, int lane) {
    int m  = row0 + (lane & 15);
    int kb = (lane < 16) ? 0 : 16;
    const half_t* q = p + m * stride + kOff + kb;
    v16h a;
#pragma unroll
    for (int j = 0; j < 8; ++j) { a[j] = q[j]; a[8 + j] = q[8 + j]; }
    return a;
}

__device__ inline v16h load_b_pk(const half_t* pw, int tile, int lane) {
    const half_t* q = pw + (size_t)tile * 512 + lane * 16;
    v16h b;
#pragma unroll
    for (int j = 0; j < 16; ++j) b[j] = q[j];
    return b;
}

// A tile with K=8 valid (rest zero): only lanes<16, halves 0..7 carry data.
__device__ inline v16h load_a_k8(const half_t* p, int row0, int stride, int cOff, int lane) {
    v16h a = {};
    if (lane < 16) {
        const half_t* q = p + (row0 + lane) * stride + cOff;
#pragma unroll
        for (int j = 0; j < 8; ++j) a[j] = q[j];
    }
    return a;
}

// B tile with K=8 valid: lane<16 holds K 0..15 -> only halves 0..7 valid.
__device__ inline v16h load_b_k8(const half_t* p, int n0, int stride, int cOff, int lane) {
    v16h b = {};
    if (lane < 16) {
        const half_t* q = p + (n0 + lane) * stride + cOff;
#pragma unroll
        for (int j = 0; j < 8; ++j) b[j] = q[j];
    }
    return b;
}

// B tile from v-transpose [64][32] LDS: col n = lane&15 (valid n<8), K 0..15
// for lane<16, K 16..31 for lane>=16.
__device__ inline v16h load_b_vt(const half_t* vtp, int cOff, int lane) {
    v16h b = {};
    int n = lane & 15;
    if (n < 8) {
        const half_t* q = vtp + (cOff + n) * 32 + ((lane < 16) ? 0 : 16);
#pragma unroll
        for (int j = 0; j < 16; ++j) b[j] = q[j];
    }
    return b;
}

// ---------------------------------------------------------------------------
// Kernel: NCHW -> NHWC transpose of x (coalesced both directions via LDS tile)
// ---------------------------------------------------------------------------
__global__ __launch_bounds__(1024)
void k_transpose(const float* __restrict__ x, float* __restrict__ xT) {
    __shared__ float t[32][33];
    int tx = threadIdx.x, ty = threadIdx.y;
    int w0 = blockIdx.x * 32, c0 = blockIdx.y * 32;
    int b  = blockIdx.z >> 6, h = blockIdx.z & 63;
    t[ty][tx] = x[(((size_t)b * 64 + (c0 + ty)) * 64 + h) * 64 + (w0 + tx)];
    __syncthreads();
    xT[(((size_t)b * 64 + h) * 64 + (w0 + ty)) * 64 + (c0 + tx)] = t[tx][ty];
}

// ---------------------------------------------------------------------------
// Kernel: pack an f32 weight matrix W[N][K] into f16 WMMA B-operand tiles.
// tile = nt*Kt + kt ; per-lane 16 contiguous halves.
// ---------------------------------------------------------------------------
__global__ __launch_bounds__(32)
void k_pack(const float* __restrict__ W, half_t* __restrict__ out, int K, int Kt) {
    int tile = blockIdx.x, lane = threadIdx.x;
    int nt = tile / Kt, kt = tile % Kt;
    int n    = nt * 16 + (lane & 15);
    int kofs = kt * 32 + ((lane < 16) ? 0 : 16);
    half_t* dst = out + (size_t)tile * 512 + lane * 16;
#pragma unroll
    for (int j = 0; j < 16; ++j) dst[j] = (half_t)W[(size_t)n * K + kofs + j];
}

// ---------------------------------------------------------------------------
// Kernel A: per-pixel windowed attention -> out_sumed [16384][64]
// One wave per pixel, 2 waves per 64-thread workgroup (disjoint LDS slices).
// ---------------------------------------------------------------------------
__global__ __launch_bounds__(64)
void k_attn(const float* __restrict__ xT,
            const float* __restrict__ n1w, const float* __restrict__ n1b,
            const float* __restrict__ qkvb, const float* __restrict__ rpb,
            const half_t* __restrict__ pw_qkv,
            float* __restrict__ outsum) {
    __shared__ half_t ln_s[2][32 * 64];
    __shared__ half_t qk_s[2][32 * 192];
    __shared__ half_t vt_s[2][64 * 32];
    __shared__ float  sc_s[2][32 * 32];
    __shared__ half_t at_s[2][32 * 32];
    __shared__ float  ss_s[2][64];

    const int wv   = threadIdx.x >> 5;
    const int lane = threadIdx.x & 31;
    half_t* ln = ln_s[wv];
    half_t* qk = qk_s[wv];
    half_t* vt = vt_s[wv];
    float*  sc = sc_s[wv];
    half_t* at = at_s[wv];
    float*  ss = ss_s[wv];

    const int pix = blockIdx.x * 2 + wv;
    const int b  = pix >> 12;
    const int h0 = (pix >> 6) & 63;
    const int w0 = pix & 63;

    const float wA = n1w[lane],      wB = n1w[lane + 32];
    const float bA = n1b[lane],      bB = n1b[lane + 32];
    float cs0 = 0.f, cs1 = 0.f;   // column sums of x_masked (residual term)

    // ---- Stage 1: masked 5x5 window, per-row layernorm (rows 0..11 live) ----
    for (int p = 0; p < 12; ++p) {
        int hh = h0 + (p / 5) - 2;
        int ww = w0 + (p % 5) - 2;
        float v0 = 0.f, v1 = 0.f;
        if (hh >= 0 && hh < 64 && ww >= 0 && ww < 64) {
            const float* px = xT + (((size_t)b * 64 + hh) * 64 + ww) * 64;
            v0 = px[lane];
            v1 = px[lane + 32];
        }
        cs0 += v0; cs1 += v1;
        float s = v0 + v1, q = v0 * v0 + v1 * v1;
        for (int off = 16; off >= 1; off >>= 1) {
            s += __shfl_xor(s, off);
            q += __shfl_xor(q, off);
        }
        float mean = s * (1.f / 64.f);
        float var  = q * (1.f / 64.f) - mean * mean;
        float rstd = rsqrtf(var + 1e-5f);
        ln[p * 64 + lane]      = (half_t)((v0 - mean) * rstd * wA + bA);
        ln[p * 64 + lane + 32] = (half_t)((v1 - mean) * rstd * wB + bB);
    }
    // masked rows: LN(0-row) == norm1_b ; pad rows 25..31 with zeros
    for (int p = 12; p < 25; ++p) {
        ln[p * 64 + lane]      = (half_t)bA;
        ln[p * 64 + lane + 32] = (half_t)bB;
    }
    for (int p = 25; p < 32; ++p) {
        ln[p * 64 + lane]      = (half_t)0.f;
        ln[p * 64 + lane + 32] = (half_t)0.f;
    }
    WAVE_LDS_FENCE();

    // ---- Stage 2: QKV = ln @ Wqkv^T + b ; q pre-scaled ; v also transposed --
    for (int mt = 0; mt < 2; ++mt) {
        v16h a0 = load_a(ln, mt * 16, 64, 0,  lane);
        v16h a1 = load_a(ln, mt * 16, 64, 32, lane);
        for (int nt = 0; nt < 12; ++nt) {
            v8f acc = {};
            acc = WMMA(a0, load_b_pk(pw_qkv, nt * 2 + 0, lane), acc);
            acc = WMMA(a1, load_b_pk(pw_qkv, nt * 2 + 1, lane), acc);
            int   n    = nt * 16 + (lane & 15);
            float bias = qkvb[n];
            float scl  = (n < 64) ? 0.35355339059327373f : 1.f;  // 1/sqrt(hd)
            int rbase = mt * 16 + ((lane >> 4) & 1) * 8;
#pragma unroll
            for (int r = 0; r < 8; ++r) {
                half_t hv = (half_t)((acc[r] + bias) * scl);
                qk[(rbase + r) * 192 + n] = hv;
                if (n >= 128) vt[(n - 128) * 32 + (rbase + r)] = hv;  // v^T
            }
        }
    }
    WAVE_LDS_FENCE();

    // ---- Stage 3: per-head attention ----
    for (int hd = 0; hd < 8; ++hd) {
        // scores = q @ k^T (+ rel-pos bias)
        for (int mt = 0; mt < 2; ++mt) {
            v16h aq = load_a_k8(qk, mt * 16, 192, hd * 8, lane);
            for (int nt = 0; nt < 2; ++nt) {
                v16h bk = load_b_k8(qk, nt * 16, 192, 64 + hd * 8, lane);
                v8f acc = {};
                acc = WMMA(aq, bk, acc);
                int n = nt * 16 + (lane & 15);
                int rbase = mt * 16 + ((lane >> 4) & 1) * 8;
#pragma unroll
                for (int r = 0; r < 8; ++r) {
                    int m = rbase + r;
                    float v = acc[r];
                    if (m < 25 && n < 25) {
                        int idx = (m / 5 - n / 5 + 4) * 9 + (m % 5 - n % 5 + 4);
                        v += rpb[idx * 8 + hd];
                    }
                    sc[m * 32 + n] = v;
                }
            }
        }
        WAVE_LDS_FENCE();

        // softmax: one row per lane, row held in registers (single LDS pass)
        {
            int m = lane;
            if (m < 25) {
                float e[25];
                float mx = -1e30f;
#pragma unroll
                for (int n = 0; n < 25; ++n) {
                    e[n] = sc[m * 32 + n];
                    mx = fmaxf(mx, e[n]);
                }
                float den = 0.f;
#pragma unroll
                for (int n = 0; n < 25; ++n) {
                    e[n] = __expf(e[n] - mx);
                    den += e[n];
                }
                float inv = 1.f / den;
#pragma unroll
                for (int n = 0; n < 25; ++n) at[m * 32 + n] = (half_t)(e[n] * inv);
#pragma unroll
                for (int n = 25; n < 32; ++n) at[m * 32 + n] = (half_t)0.f;
            } else {
#pragma unroll
                for (int n = 0; n < 32; ++n) at[m * 32 + n] = (half_t)0.f;
            }
        }
        WAVE_LDS_FENCE();

        // o_h = attn @ v_h ; fold straight into column sums over rows < 25
        v16h bv = load_b_vt(vt, hd * 8, lane);
        float psum = 0.f;
        for (int mt = 0; mt < 2; ++mt) {
            v16h aa = load_a(at, mt * 16, 32, 0, lane);
            v8f acc = {};
            acc = WMMA(aa, bv, acc);
            int rbase = mt * 16 + ((lane >> 4) & 1) * 8;
#pragma unroll
            for (int r = 0; r < 8; ++r)
                if (rbase + r < 25) psum += acc[r];
        }
        psum += __shfl_xor(psum, 16);      // combine row halves (same column)
        if (lane < 8) ss[hd * 8 + lane] = psum;
        WAVE_LDS_FENCE();
    }

    outsum[(size_t)pix * 64 + lane]      = ss[lane]      + cs0;
    outsum[(size_t)pix * 64 + lane + 32] = ss[lane + 32] + cs1;
}

// ---------------------------------------------------------------------------
// Kernel B: fused LN2 + MLP(GELU) + residual + proj, 16 pixel-rows per wave.
// ---------------------------------------------------------------------------
__global__ __launch_bounds__(32)
void k_mlp(const float* __restrict__ sumin,
           const float* __restrict__ n2w, const float* __restrict__ n2b,
           const float* __restrict__ b1,  const float* __restrict__ b2,
           const float* __restrict__ pb,
           const half_t* __restrict__ pw1, const half_t* __restrict__ pw2,
           const half_t* __restrict__ pwp,
           float* __restrict__ out) {
    __shared__ float  s_f[16 * 64];
    __shared__ half_t y_h[16 * 64];
    __shared__ half_t h1_h[16 * 256];
    __shared__ half_t y2_h[16 * 64];
    __shared__ float  o_f[16 * 64];

    const int lane = threadIdx.x;
    const int pix0 = blockIdx.x * 16;
    const float wA = n2w[lane], wB = n2w[lane + 32];
    const float bA = n2b[lane], bB = n2b[lane + 32];

    // LN2 (row per iteration, shuffle reduce)
    for (int r = 0; r < 16; ++r) {
        float v0 = sumin[(size_t)(pix0 + r) * 64 + lane];
        float v1 = sumin[(size_t)(pix0 + r) * 64 + lane + 32];
        s_f[r * 64 + lane]      = v0;
        s_f[r * 64 + lane + 32] = v1;
        float s = v0 + v1, q = v0 * v0 + v1 * v1;
        for (int off = 16; off >= 1; off >>= 1) {
            s += __shfl_xor(s, off);
            q += __shfl_xor(q, off);
        }
        float mean = s * (1.f / 64.f);
        float var  = q * (1.f / 64.f) - mean * mean;
        float rstd = rsqrtf(var + 1e-5f);
        y_h[r * 64 + lane]      = (half_t)((v0 - mean) * rstd * wA + bA);
        y_h[r * 64 + lane + 32] = (half_t)((v1 - mean) * rstd * wB + bB);
    }
    WAVE_LDS_FENCE();

    // fc1 + exact GELU
    {
        v16h a0 = load_a(y_h, 0, 64, 0,  lane);
        v16h a1 = load_a(y_h, 0, 64, 32, lane);
        for (int nt = 0; nt < 16; ++nt) {
            v8f acc = {};
            acc = WMMA(a0, load_b_pk(pw1, nt * 2 + 0, lane), acc);
            acc = WMMA(a1, load_b_pk(pw1, nt * 2 + 1, lane), acc);
            int n = nt * 16 + (lane & 15);
            float bias = b1[n];
            int rb = ((lane >> 4) & 1) * 8;
#pragma unroll
            for (int r = 0; r < 8; ++r) {
                float x = acc[r] + bias;
                float g = 0.5f * x * (1.f + erff(x * 0.70710678118654752f));
                h1_h[(rb + r) * 256 + n] = (half_t)g;
            }
        }
    }
    WAVE_LDS_FENCE();

    // fc2 + residual (hoist the 8 A-tiles once)
    {
        v16h ah[8];
#pragma unroll
        for (int kt = 0; kt < 8; ++kt) ah[kt] = load_a(h1_h, 0, 256, kt * 32, lane);
        for (int nt = 0; nt < 4; ++nt) {
            v8f acc = {};
#pragma unroll
            for (int kt = 0; kt < 8; ++kt)
                acc = WMMA(ah[kt], load_b_pk(pw2, nt * 8 + kt, lane), acc);
            int n = nt * 16 + (lane & 15);
            float bias = b2[n];
            int rb = ((lane >> 4) & 1) * 8;
#pragma unroll
            for (int r = 0; r < 8; ++r) {
                float v = acc[r] + bias + s_f[(rb + r) * 64 + n];
                y2_h[(rb + r) * 64 + n] = (half_t)v;
            }
        }
    }
    WAVE_LDS_FENCE();

    // proj
    {
        v16h a0 = load_a(y2_h, 0, 64, 0,  lane);
        v16h a1 = load_a(y2_h, 0, 64, 32, lane);
        for (int nt = 0; nt < 4; ++nt) {
            v8f acc = {};
            acc = WMMA(a0, load_b_pk(pwp, nt * 2 + 0, lane), acc);
            acc = WMMA(a1, load_b_pk(pwp, nt * 2 + 1, lane), acc);
            int n = nt * 16 + (lane & 15);
            float bias = pb[n];
            int rb = ((lane >> 4) & 1) * 8;
#pragma unroll
            for (int r = 0; r < 8; ++r) o_f[(rb + r) * 64 + n] = acc[r] + bias;
        }
    }
    WAVE_LDS_FENCE();

    // write channel-major output [B,64,H,W]; 16 consecutive w per segment
    for (int it = 0; it < 32; ++it) {
        int e = it * 32 + lane;
        int c = e >> 4, r = e & 15;
        int pix = pix0 + r;
        int b = pix >> 12, hh = (pix >> 6) & 63, ww = pix & 63;
        out[(((size_t)b * 64 + c) * 64 + hh) * 64 + ww] = o_f[r * 64 + c];
    }
}

// ---------------------------------------------------------------------------
extern "C" void kernel_launch(void* const* d_in, const int* in_sizes, int n_in,
                              void* d_out, int out_size, void* d_ws, size_t ws_size,
                              hipStream_t stream) {
    const float* x    = (const float*)d_in[0];
    const float* n1w  = (const float*)d_in[1];
    const float* n1b  = (const float*)d_in[2];
    const float* qkvw = (const float*)d_in[3];
    const float* qkvb = (const float*)d_in[4];
    const float* rpb  = (const float*)d_in[5];
    const float* n2w  = (const float*)d_in[6];
    const float* n2b  = (const float*)d_in[7];
    const float* f1w  = (const float*)d_in[8];
    const float* f1b  = (const float*)d_in[9];
    const float* f2w  = (const float*)d_in[10];
    const float* f2b  = (const float*)d_in[11];
    const float* pjw  = (const float*)d_in[12];
    const float* pjb  = (const float*)d_in[13];
    float* out = (float*)d_out;

    float*  xT   = (float*)d_ws;                              // 4 MB
    float*  sums = (float*)((char*)d_ws + (4u << 20));        // 4 MB
    half_t* pk   = (half_t*)((char*)d_ws + (8u << 20));       // 96 KB packed
    half_t* pk_qkv = pk;                //  24 tiles (N=192,K=64)
    half_t* pk_fc1 = pk + 24 * 512;     //  32 tiles (N=256,K=64)
    half_t* pk_fc2 = pk + 56 * 512;     //  32 tiles (N=64,K=256)
    half_t* pk_prj = pk + 88 * 512;     //   8 tiles (N=64,K=64)

    k_pack<<<24, 32, 0, stream>>>(qkvw, pk_qkv, 64, 2);
    k_pack<<<32, 32, 0, stream>>>(f1w,  pk_fc1, 64, 2);
    k_pack<<<32, 32, 0, stream>>>(f2w,  pk_fc2, 256, 8);
    k_pack<<<8,  32, 0, stream>>>(pjw,  pk_prj, 64, 2);

    dim3 tb(32, 32), tg(2, 2, 256);
    k_transpose<<<tg, tb, 0, stream>>>(x, xT);

    k_attn<<<8192, 64, 0, stream>>>(xT, n1w, n1b, qkvb, rpb, pk_qkv, sums);
    k_mlp<<<1024, 32, 0, stream>>>(sums, n2w, n2b, f1b, f2b, pjb,
                                   pk_fc1, pk_fc2, pk_prj, out);
}
